// FeedForwardNetwork_37873021616386
// MI455X (gfx1250) — matrix operand order repaired and checked by reading, not compiled, not run on hardware
//
#include <hip/hip_runtime.h>
#include <hip/hip_bf16.h>

typedef __attribute__((ext_vector_type(16))) _Float16 v16h;
typedef __attribute__((ext_vector_type(8)))  _Float16 v8h;
typedef __attribute__((ext_vector_type(8)))  float    v8f;

#define WAVES 4

// ---- packed-weight layout in d_ws (units: halfs) ----
// Each fragment = 512 halfs: [lane 0..31][e 0..15], value = W[K = ks*32 + (e&7) + 16*(e>>3) + 8*(lane>>4)][N = nt*16 + (lane&15)]
// fragment index = nt * nks + ks
#define PK_W1_0 0        // K=128 N=672 -> 42 nt * 4 ks
#define PK_W1_1 86016    // K=64  N=192 -> 12 nt * 2 ks
#define PK_W1_2 98304    // K=32  N=96  ->  6 nt * 1 ks
#define PK_W2_0 101376   // K=384 N=128 ->  8 nt * 12 ks
#define PK_W2_1 150528   // K=192 N=64  ->  4 nt * 6 ks
#define PK_W2_2 162816   // K=96  N=32  ->  2 nt * 3 ks
#define PK_TOTAL 165888  // halfs (331776 bytes)

// ---- per-wave LDS layout (units: halfs) ----
#define L_S   0          // silu scalars    [16][384]
#define L_G   6144       // sigmoid gates   [16][288]
#define L_V1  10752      // gated l=1 acts  [3][16][192]
#define L_V2  19968      // gated l=2 acts  [5][16][96]
#define L_PER_WAVE 27648 // 55296 bytes per wave
#define LDS_BYTES (WAVES * L_PER_WAVE * 2)

__global__ void pack_weights_kernel(const float* __restrict__ W1_0, const float* __restrict__ W1_1,
                                    const float* __restrict__ W1_2, const float* __restrict__ W2_0,
                                    const float* __restrict__ W2_1, const float* __restrict__ W2_2,
                                    _Float16* __restrict__ ws) {
  int idx = blockIdx.x * blockDim.x + threadIdx.x;
  if (idx >= PK_TOTAL) return;
  const float* src; int base, nks, ncols;
  if      (idx < PK_W1_1) { src = W1_0; base = PK_W1_0; nks = 4;  ncols = 672; }
  else if (idx < PK_W1_2) { src = W1_1; base = PK_W1_1; nks = 2;  ncols = 192; }
  else if (idx < PK_W2_0) { src = W1_2; base = PK_W1_2; nks = 1;  ncols = 96;  }
  else if (idx < PK_W2_1) { src = W2_0; base = PK_W2_0; nks = 12; ncols = 128; }
  else if (idx < PK_W2_2) { src = W2_1; base = PK_W2_1; nks = 6;  ncols = 64;  }
  else                    { src = W2_2; base = PK_W2_2; nks = 3;  ncols = 32;  }
  int local = idx - base;
  int frag = local >> 9, w = local & 511;
  int lane = w >> 4, e = w & 15;
  int nt = frag / nks, ks = frag % nks;
  int K   = ks * 32 + (e & 7) + ((e >> 3) << 4) + 8 * (lane >> 4); // ISA 16-bit B layout: K = (e&7) + 16*(e>>3) + 8*hi
  int col = nt * 16 + (lane & 15);                // N = lane % 16
  ws[idx] = (_Float16)src[K * ncols + col];
}

__device__ __forceinline__ v16h wmma_b_load(const _Float16* p) {
  return *(const v16h*)p;   // 32B contiguous, lane-major pre-packed fragment
}

// A-fragment from row-major f16 LDS row: e<8 -> K = colLo+e ; e>=8 -> K = colLo+16+(e-8)
__device__ __forceinline__ v16h lds_a_load(const _Float16* row, int colLo) {
  v8h lo = *(const v8h*)(row + colLo);
  v8h hp = *(const v8h*)(row + colLo + 16);
  v16h r;
#pragma unroll
  for (int i = 0; i < 8; ++i) { r[i] = lo[i]; r[i + 8] = hp[i]; }
  return r;
}

// fast sigmoid: v_exp_f32 + v_rcp_f32 (TRANS ops co-execute with WMMA); avoids the
// IEEE v_div_scale/v_div_fmas/v_div_fixup sequence the plain division lowered to.
__device__ __forceinline__ float sigm(float x) {
  return __builtin_amdgcn_rcpf(1.0f + __expf(-x));
}

__device__ __forceinline__ v8f wmma16(v16h a, v16h b, v8f c) {
  return __builtin_amdgcn_wmma_f32_16x16x32_f16(false, a, false, b, (short)0, c, false, false);
}

__global__ __launch_bounds__(WAVES * 32)
void ffn_main_kernel(const float* __restrict__ x, const _Float16* __restrict__ wpk,
                     const float* __restrict__ b1, const float* __restrict__ b2,
                     float* __restrict__ out, int nrows, int ntiles) {
  extern __shared__ _Float16 smem[];
  const int wave = threadIdx.x >> 5;
  const int lane = threadIdx.x & 31;
  const int tile = blockIdx.x * WAVES + wave;
  if (tile >= ntiles) return;                     // wave-uniform exit, EXEC stays full otherwise
  const int rl = lane & 15;                       // A row / C column within 16x16 tile
  const int hi = lane >> 4;
  _Float16* ls = smem + wave * L_PER_WAVE;

  // ---------- Phase 0: load input rows, convert to f16 A-fragments ----------
  int arow = tile * 16 + rl; if (arow >= nrows) arow = nrows - 1;
  const float* xr = x + (size_t)arow * 480;
  v16h a0[4], a1[3][2], a2[5];
#pragma unroll
  for (int ks = 0; ks < 4; ++ks)
#pragma unroll
    for (int e = 0; e < 16; ++e) {
      int k = (e & 7) + ((e >> 3) << 4) + 8 * hi; // ISA 16-bit A layout
      a0[ks][e] = (_Float16)xr[ks * 32 + k];
    }
#pragma unroll
  for (int m = 0; m < 3; ++m)
#pragma unroll
    for (int ks = 0; ks < 2; ++ks)
#pragma unroll
      for (int e = 0; e < 16; ++e) {
        int k = (e & 7) + ((e >> 3) << 4) + 8 * hi;
        a1[m][ks][e] = (_Float16)xr[128 + 3 * (ks * 32 + k) + m];
      }
#pragma unroll
  for (int m = 0; m < 5; ++m)
#pragma unroll
    for (int e = 0; e < 16; ++e) {
      int k = (e & 7) + ((e >> 3) << 4) + 8 * hi;
      a2[m][e] = (_Float16)xr[320 + 5 * k + m];
    }

  // ---------- Phase 1a: s_all = x0 @ W1_0 / sqrt(128) + b1; SiLU / sigmoid ----------
  {
    const _Float16* wp = wpk + PK_W1_0 + lane * 16;
    for (int nt = 0; nt < 42; ++nt) {
      v8f c = {};
#pragma unroll
      for (int ks = 0; ks < 4; ++ks)
        c = wmma16(a0[ks], wmma_b_load(wp + (nt * 4 + ks) * 512), c);
      const int col = nt * 16 + rl;
      const float bias = b1[col];
      if (nt < 24) {
#pragma unroll
        for (int r = 0; r < 8; ++r) {
          float v = c[r] * 0.08838834764831845f + bias;      // 1/sqrt(128)
          ls[L_S + (r + 8 * hi) * 384 + col] = (_Float16)(v * sigm(v));
        }
      } else {
#pragma unroll
        for (int r = 0; r < 8; ++r) {
          float v = c[r] * 0.08838834764831845f + bias;
          ls[L_G + (r + 8 * hi) * 288 + (col - 384)] = (_Float16)sigm(v);
        }
      }
    }
  }

  // ---------- Phase 1b: v1 = (x1 @ W1_1)/sqrt(64) * gate ----------
  {
    const _Float16* wp = wpk + PK_W1_1 + lane * 16;
    for (int m = 0; m < 3; ++m)
      for (int nt = 0; nt < 12; ++nt) {
        v8f c = {};
#pragma unroll
        for (int ks = 0; ks < 2; ++ks)
          c = wmma16(a1[m][ks], wmma_b_load(wp + (nt * 2 + ks) * 512), c);
        int gc = nt * 16 + rl;
#pragma unroll
        for (int r = 0; r < 8; ++r) {
          int row = r + 8 * hi;
          float gv = (float)ls[L_G + row * 288 + gc];
          ls[L_V1 + m * 3072 + row * 192 + gc] = (_Float16)(c[r] * 0.125f * gv); // 1/sqrt(64)
        }
      }
  }

  // ---------- Phase 1c: v2 = (x2 @ W1_2)/sqrt(32) * gate ----------
  {
    const _Float16* wp = wpk + PK_W1_2 + lane * 16;
    for (int m = 0; m < 5; ++m)
      for (int nt = 0; nt < 6; ++nt) {
        v8f c = {};
        c = wmma16(a2[m], wmma_b_load(wp + nt * 512), c);
        int gc = nt * 16 + rl;
#pragma unroll
        for (int r = 0; r < 8; ++r) {
          int row = r + 8 * hi;
          float gv = (float)ls[L_G + row * 288 + 192 + gc];
          ls[L_V2 + m * 1536 + row * 96 + gc] = (_Float16)(c[r] * 0.17677669529663687f * gv); // 1/sqrt(32)
        }
      }
  }

  // ---------- Phase 2a: y0 = s @ W2_0 / sqrt(384) + b2 ----------
  {
    const _Float16* srow = ls + L_S + rl * 384;
    v16h sA[12];
#pragma unroll
    for (int ks = 0; ks < 12; ++ks) sA[ks] = lds_a_load(srow, ks * 32 + 8 * hi);
    const _Float16* wp = wpk + PK_W2_0 + lane * 16;
    for (int nt = 0; nt < 8; ++nt) {
      v8f c = {};
#pragma unroll
      for (int ks = 0; ks < 12; ++ks)
        c = wmma16(sA[ks], wmma_b_load(wp + (nt * 12 + ks) * 512), c);
      int col = nt * 16 + rl;
      float bias = b2[col];
#pragma unroll
      for (int r = 0; r < 8; ++r) {
        int row = tile * 16 + r + 8 * hi;
        if (row < nrows)
          out[(size_t)row * 480 + col] = c[r] * 0.05103103630798288f + bias; // 1/sqrt(384)
      }
    }
  }

  // ---------- Phase 2b: y1 = v1 @ W2_1 / sqrt(192) ----------
  for (int m = 0; m < 3; ++m) {
    const _Float16* vrow = ls + L_V1 + m * 3072 + rl * 192;
    v16h vA[6];
#pragma unroll
    for (int ks = 0; ks < 6; ++ks) vA[ks] = lds_a_load(vrow, ks * 32 + 8 * hi);
    const _Float16* wp = wpk + PK_W2_1 + lane * 16;
    for (int nt = 0; nt < 4; ++nt) {
      v8f c = {};
#pragma unroll
      for (int ks = 0; ks < 6; ++ks)
        c = wmma16(vA[ks], wmma_b_load(wp + (nt * 6 + ks) * 512), c);
      int v = nt * 16 + rl;
      int col = 128 + 3 * v + m;
#pragma unroll
      for (int r = 0; r < 8; ++r) {
        int row = tile * 16 + r + 8 * hi;
        if (row < nrows)
          out[(size_t)row * 480 + col] = c[r] * 0.07216878364870323f; // 1/sqrt(192)
      }
    }
  }

  // ---------- Phase 2c: y2 = v2 @ W2_2 / sqrt(96) ----------
  for (int m = 0; m < 5; ++m) {
    const _Float16* vrow = ls + L_V2 + m * 1536 + rl * 96;
    v16h vA[3];
#pragma unroll
    for (int ks = 0; ks < 3; ++ks) vA[ks] = lds_a_load(vrow, ks * 32 + 8 * hi);
    const _Float16* wp = wpk + PK_W2_2 + lane * 16;
    for (int nt = 0; nt < 2; ++nt) {
      v8f c = {};
#pragma unroll
      for (int ks = 0; ks < 3; ++ks)
        c = wmma16(vA[ks], wmma_b_load(wp + (nt * 3 + ks) * 512), c);
      int v = nt * 16 + rl;
      int col = 320 + 5 * v + m;
#pragma unroll
      for (int r = 0; r < 8; ++r) {
        int row = tile * 16 + r + 8 * hi;
        if (row < nrows)
          out[(size_t)row * 480 + col] = c[r] * 0.10206207261596577f; // 1/sqrt(96)
      }
    }
  }
}

extern "C" void kernel_launch(void* const* d_in, const int* in_sizes, int n_in,
                              void* d_out, int out_size, void* d_ws, size_t ws_size,
                              hipStream_t stream) {
  const float* x    = (const float*)d_in[0];
  const float* W1_0 = (const float*)d_in[1];
  const float* W1_1 = (const float*)d_in[2];
  const float* W1_2 = (const float*)d_in[3];
  const float* b1   = (const float*)d_in[4];
  const float* W2_0 = (const float*)d_in[5];
  const float* W2_1 = (const float*)d_in[6];
  const float* W2_2 = (const float*)d_in[7];
  const float* b2   = (const float*)d_in[8];
  float* out = (float*)d_out;
  _Float16* wpk = (_Float16*)d_ws;

  int nrows  = in_sizes[0] / 480;
  int ntiles = (nrows + 15) / 16;

  pack_weights_kernel<<<(PK_TOTAL + 255) / 256, 256, 0, stream>>>(
      W1_0, W1_1, W1_2, W2_0, W2_1, W2_2, wpk);

  int blocks = (ntiles + WAVES - 1) / WAVES;
  ffn_main_kernel<<<blocks, WAVES * 32, LDS_BYTES, stream>>>(
      x, wpk, b1, b2, out, nrows, ntiles);
}
